// PairwiseInteractionHead_10144712753220
// MI455X (gfx1250) — matrix operand order repaired and checked by reading, not compiled, not run on hardware
//
#include <hip/hip_runtime.h>
#include <hip/hip_bf16.h>
#include <math.h>

typedef __attribute__((ext_vector_type(16))) _Float16 v16h;
typedef __attribute__((ext_vector_type(8)))  _Float16 v8h;
typedef __attribute__((ext_vector_type(8)))  float    v8f;

namespace {
constexpr int B_   = 4;
constexpr int LL   = 256;
constexpr int LP   = 256;
constexpr int D    = 256;
constexpr int H    = 128;
constexpr int TOPK = 100;
constexpr int F4D  = 4 * D;   // 1024
}

__device__ __forceinline__ float gelu_erf(float x) {
  return 0.5f * x * (1.0f + erff(x * 0.70710678118654752f));
}

// out[row,e] = bias[e] + sum_k tok[row,k] * W[e,k]      (row = b*L + l)
__global__ __launch_bounds__(256) void proj_kernel(const float* __restrict__ tok,
                                                   const float* __restrict__ W,
                                                   const float* __restrict__ bias,
                                                   float* __restrict__ out) {
  __shared__ float xs[D];
  const int row = blockIdx.x;
  const int e   = threadIdx.x;                // 0..255
  xs[e] = tok[row * D + e];
  __syncthreads();
  float acc = bias[e];
  const float* wr = W + (size_t)e * D;
#pragma unroll 8
  for (int k = 0; k < D; ++k) acc = fmaf(xs[k], wr[k], acc);
  out[row * D + e] = acc;
}

// A-term: out[row,h] = (addBias? b1[h] : 0) + sum_k W1[h, colOff+k] * x[row,k]
__global__ __launch_bounds__(128) void aterm_kernel(const float* __restrict__ x,
                                                    const float* __restrict__ W1,
                                                    const float* __restrict__ b1,
                                                    int colOff, int addBias,
                                                    float* __restrict__ out) {
  __shared__ float xs[D];
  const int row = blockIdx.x;
  const int t   = threadIdx.x;                // 0..127
  xs[t]       = x[row * D + t];
  xs[t + 128] = x[row * D + t + 128];
  __syncthreads();
  float acc = addBias ? b1[t] : 0.0f;
  const float* wr = W1 + (size_t)t * F4D + colOff;
#pragma unroll 8
  for (int k = 0; k < D; ++k) acc = fmaf(xs[k], wr[k], acc);
  out[row * H + t] = acc;
}

// f32 -> f16 weight conversion: W1c = W1[:,2D:3D], W1d = W1[:,3D:4D], W2f = W2
__global__ __launch_bounds__(256) void cvt_kernel(const float* __restrict__ W1,
                                                  const float* __restrict__ W2,
                                                  _Float16* __restrict__ w1c,
                                                  _Float16* __restrict__ w1d,
                                                  _Float16* __restrict__ w2f) {
  const int i = blockIdx.x * 256 + threadIdx.x;
  if (i < H * D) {
    const int h = i >> 8, k = i & 255;
    w1c[i] = (_Float16)W1[(size_t)h * F4D + 2 * D + k];
    w1d[i] = (_Float16)W1[(size_t)h * F4D + 3 * D + k];
  }
  if (i < H * H) w2f[i] = (_Float16)W2[i];
}

// Main per-pair MLP.  One block = (b, i, tile of 32 j's) = two 16-row M tiles.
// 8 waves, each owns a 16-wide slab of H.  Every B (weight) fragment is loaded
// once from global and consumed by TWO WMMAs (one per M tile), halving L2
// weight traffic per MAC.  Layer1: K=512 (prod||absdiff), accumulator
// preloaded with A1[i]+A2[j]+b1.  Layer2: K=128.
__global__ __launch_bounds__(256) void pair_kernel(
    const float* __restrict__ l_, const float* __restrict__ p_,
    const float* __restrict__ A1, const float* __restrict__ A2,
    const _Float16* __restrict__ W1c, const _Float16* __restrict__ W1d,
    const _Float16* __restrict__ W2f,
    const float* __restrict__ b2, const float* __restrict__ W3,
    const float* __restrict__ b3,
    const unsigned char* __restrict__ l_pad, const unsigned char* __restrict__ p_pad,
    float* __restrict__ pair_logit, float* __restrict__ pair_prob,
    float* __restrict__ scratch) {
  // Row strides padded (1040B / 272B / 528B) so the 16 rows of an A-fragment
  // load land on 16 distinct 4-bank groups -> minimum 2-cycle LDS access.
  __shared__ __align__(32) _Float16 feats[32][520];  // 32 pairs x 512 K (+8 pad)
  __shared__ __align__(32) _Float16 h1s[32][136];    // 32 pairs x 128 H (+8 pad)
  __shared__ __align__(32) float    h2s[32][132];    // 32 pairs x 128 H (+4 pad)

  const int b  = blockIdx.z;
  const int i  = blockIdx.y;
  const int j0 = blockIdx.x * 32;
  const int tid = threadIdx.x;

  // ---- Stage 1: build per-pair features once, share across all 8 waves ----
  const float* lrow  = l_ + ((size_t)(b * LL + i)) * D;
  const float* pbase = p_ + ((size_t)(b * LP + j0)) * D;
  const float  lk = lrow[tid];                       // k = tid (0..255)
#pragma unroll 4
  for (int m = 0; m < 32; ++m) {
    const float pk = pbase[m * D + tid];
    feats[m][tid]     = (_Float16)(lk * pk);
    feats[m][D + tid] = (_Float16)fabsf(lk - pk);
  }
  __syncthreads();

  const int wv     = tid >> 5;
  const int lane   = tid & 31;
  const int laneHi = lane >> 4;
  const int n      = lane & 15;
  const int hb     = wv * 16 + n;                    // this lane's H column

  // ---- Layer 1: two M tiles, C preloaded with A1[i]+A2[j] (b1 in A1) ----
  v8f acc0, acc1;
  const float a1v = A1[((size_t)(b * LL + i)) * H + hb];
  const float* a2p = A2 + ((size_t)(b * LP + j0)) * H + hb;
#pragma unroll
  for (int r = 0; r < 8; ++r) {
    const int m = r + laneHi * 8;
    acc0[r] = a1v + a2p[(size_t)m * H];
    acc1[r] = a1v + a2p[(size_t)(m + 16) * H];
  }
#pragma unroll
  for (int ks = 0; ks < 16; ++ks) {
    // B fragment: column n = W1 row hb, 16 contiguous f16 (32B) per lane
    const _Float16* wsrc = (ks < 8)
        ? (W1c + (size_t)hb * D + ks * 32 + laneHi * 16)
        : (W1d + (size_t)hb * D + (ks - 8) * 32 + laneHi * 16);
    const v16h bf = *(const v16h*)wsrc;
    // A fragments: 16 contiguous k per lane (k = 16*laneHi + s), two 16B chunks
    const int c0 = ks * 32 + laneHi * 16;
    union { v16h v; v8h h[2]; } af0, af1;
    af0.h[0] = *(const v8h*)&feats[n][c0];
    af0.h[1] = *(const v8h*)&feats[n][c0 + 8];
    af1.h[0] = *(const v8h*)&feats[16 + n][c0];
    af1.h[1] = *(const v8h*)&feats[16 + n][c0 + 8];
    acc0 = __builtin_amdgcn_wmma_f32_16x16x32_f16(false, af0.v, false, bf,
                                                  (short)0, acc0, false, false);
    acc1 = __builtin_amdgcn_wmma_f32_16x16x32_f16(false, af1.v, false, bf,
                                                  (short)0, acc1, false, false);
  }
#pragma unroll
  for (int r = 0; r < 8; ++r) {
    const int m = r + laneHi * 8;
    h1s[m][hb]      = (_Float16)gelu_erf(acc0[r]);
    h1s[m + 16][hb] = (_Float16)gelu_erf(acc1[r]);
  }
  __syncthreads();

  // ---- Layer 2: h2 = GELU(h1 @ W2^T + b2), K = 128, two M tiles ----
  v8f acc2a, acc2b;
  const float b2v = b2[hb];
#pragma unroll
  for (int r = 0; r < 8; ++r) { acc2a[r] = b2v; acc2b[r] = b2v; }
#pragma unroll
  for (int ks = 0; ks < 4; ++ks) {
    const v16h bf = *(const v16h*)(W2f + (size_t)hb * H + ks * 32 + laneHi * 16);
    const int c0 = ks * 32 + laneHi * 16;
    union { v16h v; v8h h[2]; } af0, af1;
    af0.h[0] = *(const v8h*)&h1s[n][c0];
    af0.h[1] = *(const v8h*)&h1s[n][c0 + 8];
    af1.h[0] = *(const v8h*)&h1s[16 + n][c0];
    af1.h[1] = *(const v8h*)&h1s[16 + n][c0 + 8];
    acc2a = __builtin_amdgcn_wmma_f32_16x16x32_f16(false, af0.v, false, bf,
                                                   (short)0, acc2a, false, false);
    acc2b = __builtin_amdgcn_wmma_f32_16x16x32_f16(false, af1.v, false, bf,
                                                   (short)0, acc2b, false, false);
  }
#pragma unroll
  for (int r = 0; r < 8; ++r) {
    const int m = r + laneHi * 8;
    h2s[m][hb]      = gelu_erf(acc2a[r]);
    h2s[m + 16][hb] = gelu_erf(acc2b[r]);
  }
  __syncthreads();

  // ---- Layer 3 + epilogue: each wave finishes 4 of the 32 pairs ----
  const float b3v = b3[0];
  const bool li_bad = (l_pad[b * LL + i] != 0);
#pragma unroll
  for (int mm = 0; mm < 4; ++mm) {
    const int m = wv * 4 + mm;
    float s = 0.0f;
#pragma unroll
    for (int t = 0; t < 4; ++t) {
      const int h = lane + t * 32;
      s += W3[h] * h2s[m][h];
    }
#pragma unroll
    for (int off = 16; off > 0; off >>= 1) s += __shfl_down(s, off, 32);
    if (lane == 0) {
      float x = s + b3v;
      if (isnan(x))      x = 0.0f;
      else if (isinf(x)) x = (x > 0.0f) ? 20.0f : -20.0f;
      const int j = j0 + m;
      if (li_bad || (p_pad[b * LP + j] != 0)) x = -20.0f;
      const size_t idx = ((size_t)(b * LL + i)) * LP + j;
      pair_logit[idx] = x;
      pair_prob[idx]  = 1.0f / (1.0f + expf(-x));
      scratch[idx]    = x;
    }
  }
}

// Per batch: 100 destructive argmax extractions, then softmax-weighted sum.
__global__ __launch_bounds__(256) void topk_kernel(float* __restrict__ scratch,
                                                   float* __restrict__ out_logit) {
  __shared__ float sval[256];
  __shared__ int   sidx[256];
  __shared__ float topv[TOPK];
  const int b = blockIdx.x, t = threadIdx.x;
  float* base = scratch + (size_t)b * LL * LP;
  const int N = LL * LP;
  for (int it = 0; it < TOPK; ++it) {
    float best = -3.4e38f; int bi = 0;
    for (int k = t; k < N; k += 256) {
      const float v = base[k];
      if (v > best) { best = v; bi = k; }
    }
    sval[t] = best; sidx[t] = bi;
    __syncthreads();
    for (int off = 128; off > 0; off >>= 1) {
      if (t < off && sval[t + off] > sval[t]) {
        sval[t] = sval[t + off]; sidx[t] = sidx[t + off];
      }
      __syncthreads();
    }
    if (t == 0) { topv[it] = sval[0]; base[sidx[0]] = -3.4e38f; }
    __syncthreads();
  }
  if (t == 0) {
    const float mx = topv[0];     // extracted in descending order
    float sw = 0.0f, swv = 0.0f;
    for (int k = 0; k < TOPK; ++k) {
      const float w = expf(topv[k] - mx);
      sw += w; swv += w * topv[k];
    }
    out_logit[b] = swv / sw;
  }
}

extern "C" void kernel_launch(void* const* d_in, const int* in_sizes, int n_in,
                              void* d_out, int out_size, void* d_ws, size_t ws_size,
                              hipStream_t stream) {
  (void)in_sizes; (void)n_in; (void)out_size; (void)ws_size;
  const float* l_tok = (const float*)d_in[0];
  const float* p_tok = (const float*)d_in[1];
  const unsigned char* l_pad = (const unsigned char*)d_in[2];  // bool mask
  const unsigned char* p_pad = (const unsigned char*)d_in[3];
  const float* Wl = (const float*)d_in[4];
  const float* bl = (const float*)d_in[5];
  const float* Wp = (const float*)d_in[6];
  const float* bp = (const float*)d_in[7];
  const float* W1 = (const float*)d_in[8];
  const float* b1 = (const float*)d_in[9];
  const float* W2 = (const float*)d_in[10];
  const float* b2 = (const float*)d_in[11];
  const float* W3 = (const float*)d_in[12];
  const float* b3 = (const float*)d_in[13];

  // Workspace layout (all regions well-aligned by construction)
  char* ws = (char*)d_ws;
  float*    l_f   = (float*)ws;                 ws += (size_t)B_ * LL * D * 4;   // 1 MB
  float*    p_f   = (float*)ws;                 ws += (size_t)B_ * LP * D * 4;   // 1 MB
  float*    A1    = (float*)ws;                 ws += (size_t)B_ * LL * H * 4;   // 512 KB
  float*    A2    = (float*)ws;                 ws += (size_t)B_ * LP * H * 4;   // 512 KB
  _Float16* W1c   = (_Float16*)ws;              ws += (size_t)H * D * 2;         // 64 KB
  _Float16* W1d   = (_Float16*)ws;              ws += (size_t)H * D * 2;         // 64 KB
  _Float16* W2f   = (_Float16*)ws;              ws += (size_t)H * H * 2;         // 32 KB
  float*    scr   = (float*)ws;                                                  // 1 MB

  float* out_logit  = (float*)d_out;            // [B]
  float* pair_logit = out_logit + B_;           // [B,LL,LP]
  float* pair_prob  = pair_logit + (size_t)B_ * LL * LP;

  proj_kernel<<<B_ * LL, 256, 0, stream>>>(l_tok, Wl, bl, l_f);
  proj_kernel<<<B_ * LP, 256, 0, stream>>>(p_tok, Wp, bp, p_f);
  aterm_kernel<<<B_ * LL, 128, 0, stream>>>(l_f, W1, b1, 0, 1, A1);   // b1 folded here
  aterm_kernel<<<B_ * LP, 128, 0, stream>>>(p_f, W1, b1, D, 0, A2);
  cvt_kernel<<<(H * D + 255) / 256, 256, 0, stream>>>(W1, W2, W1c, W1d, W2f);

  pair_kernel<<<dim3(LP / 32, LL, B_), 256, 0, stream>>>(
      l_f, p_f, A1, A2, W1c, W1d, W2f, b2, W3, b3, l_pad, p_pad,
      pair_logit, pair_prob, scr);

  topk_kernel<<<B_, 256, 0, stream>>>(scr, out_logit);
}
